// MultiHeadAttention_78460462563626
// MI455X (gfx1250) — matrix operand order repaired and checked by reading, not compiled, not run on hardware
//
#include <hip/hip_runtime.h>
#include <stdint.h>

// ---------------- problem constants (match reference) ----------------
constexpr int Bsz = 2;
constexpr int Ssz = 2048;
constexpr int Psz = 2048;
constexpr int Esz = 1024;
constexpr int Hh  = 16;
constexpr int DHd = 64;               // E / H
constexpr int Tt  = Psz + Ssz;        // 4096
constexpr int Mrows = Bsz * Ssz;      // 4096 rows for the projection GEMMs

typedef float    v8f   __attribute__((ext_vector_type(8)));
typedef __bf16   v16bf __attribute__((ext_vector_type(16)));
typedef unsigned u32x4 __attribute__((ext_vector_type(4)));
typedef int      i32x4 __attribute__((ext_vector_type(4)));
typedef int      i32x8 __attribute__((ext_vector_type(8)));

// Make TDM writes to LDS visible to the compiler: escape the LDS base
// pointer through a memory-clobbering asm so alias analysis cannot fold
// loads of "never-written" shared memory to undef.
__device__ __forceinline__ void lds_publish(void* p) {
  asm volatile("" : "+v"(p) : : "memory");
}

// ---------------- Tensor Data Mover: 2D tile global -> LDS -----------
// Builds a D# per CDNA5 ISA (CDNA5 ISA) and issues
// TENSOR_LOAD_TO_LDS. rows x row_elems bf16 tile, source row stride
// row_stride_elems; LDS padding (pad codes per ISA: interval v ->
// 2^(v+1) dwords, amount v -> v+1 dwords) reproduces our padded LDS
// strides with zero shader-side staging work.
__device__ __forceinline__ void tdm_load_2d(uint32_t lds_off_bytes,
                                            const void* gsrc,
                                            uint32_t rows, uint32_t row_elems,
                                            uint32_t row_stride_elems,
                                            uint32_t pad_int_code,
                                            uint32_t pad_amt_code) {
  const uint64_t ga = (uint64_t)(uintptr_t)gsrc;
  u32x4 g0;
  g0[0] = 1u;                                            // count=1 (valid user D#)
  g0[1] = lds_off_bytes;                                 // lds_addr
  g0[2] = (uint32_t)(ga & 0xffffffffu);                  // global_addr[31:0]
  g0[3] = (uint32_t)((ga >> 32) & 0x01ffffffu)           // global_addr[56:32]
          | (2u << 30);                                  // type=2 ("image")

  i32x8 g1;
  g1[0] = (int)((1u << 16)                               // data_size=1 -> 2 bytes
                | (1u << 20)                             // pad_enable
                | (pad_int_code << 22)                   // pad_interval
                | (pad_amt_code << 25));                 // pad_amount
  g1[1] = (int)((row_stride_elems & 0xffffu) << 16);     // tensor_dim0[15:0]
  g1[2] = (int)(((row_stride_elems >> 16) & 0xffffu)     // tensor_dim0[31:16]
                | ((rows & 0xffffu) << 16));             // tensor_dim1[15:0]
  g1[3] = (int)(((rows >> 16) & 0xffffu)                 // tensor_dim1[31:16]
                | ((row_elems & 0xffffu) << 16));        // tile_dim0
  g1[4] = (int)(rows & 0xffffu);                         // tile_dim1 (tile_dim2=0)
  g1[5] = (int)row_stride_elems;                         // tensor_dim0_stride[31:0]
  g1[6] = 0;                                             // stride hi / dim1_stride lo
  g1[7] = 0;

  const i32x4 z4 = {0, 0, 0, 0};
#if defined(__clang_major__) && (__clang_major__ >= 23)
  const i32x8 z8 = {0, 0, 0, 0, 0, 0, 0, 0};
  __builtin_amdgcn_tensor_load_to_lds(g0, g1, z4, z4, z8, 0);
#else
  __builtin_amdgcn_tensor_load_to_lds(g0, g1, z4, z4, 0);
#endif
}

// ---------------- WMMA fragment loaders (wave32, contiguous k order) ------
// A-matrix 16x32 bf16: lanes 0-15 -> M, K 0..15; lanes 16-31 -> K 16..31.
__device__ __forceinline__ v16bf load_A_frag(const __bf16* src, int ld) {
  const int lane  = threadIdx.x & 31;
  const int m     = lane & 15;
  const int kbase = (lane >> 4) ? 16 : 0;
  v16bf a;
#pragma unroll
  for (int j = 0; j < 4; ++j) {
    const int k = kbase + 2 * j;
    a[2 * j]     = src[m * ld + k];
    a[2 * j + 1] = src[m * ld + k + 1];
  }
#pragma unroll
  for (int j = 4; j < 8; ++j) {
    const int k = kbase + 2 * j;
    a[2 * j]     = src[m * ld + k];
    a[2 * j + 1] = src[m * ld + k + 1];
  }
  return a;
}

// B-matrix 32x16 bf16, source stored [N][K] row-major (e.g. W[n][k]).
__device__ __forceinline__ v16bf load_B_frag_nk(const __bf16* src, int ld) {
  const int lane  = threadIdx.x & 31;
  const int n     = lane & 15;
  const int kbase = (lane >> 4) ? 16 : 0;
  v16bf b;
#pragma unroll
  for (int e = 0; e < 16; ++e) b[e] = src[n * ld + kbase + e];
  return b;
}

// B-matrix 32x16 bf16, source stored [K][N] row-major (e.g. V[key][dh]).
__device__ __forceinline__ v16bf load_B_frag_kn(const __bf16* src, int ld) {
  const int lane  = threadIdx.x & 31;
  const int n     = lane & 15;
  const int kbase = (lane >> 4) ? 16 : 0;
  v16bf b;
#pragma unroll
  for (int e = 0; e < 16; ++e) b[e] = src[(kbase + e) * ld + n];
  return b;
}

// ---------------- elementwise casts ----------------------------------
__global__ void cast_f32_to_bf16(const float* __restrict__ in,
                                 __bf16* __restrict__ out, int n) {
  int i = blockIdx.x * blockDim.x + threadIdx.x;
  if (i < n) out[i] = (__bf16)in[i];
}

// past_[kv] f32 [B,H,P,DH] -> bf16 cache [B,H,T,DH] positions 0..P-1
__global__ void scatter_past(const float* __restrict__ past,
                             __bf16* __restrict__ cache, int n) {
  int idx = blockIdx.x * blockDim.x + threadIdx.x;
  if (idx >= n) return;
  int d   = idx & (DHd - 1);
  int rem = idx >> 6;               // bh*P + p
  int p   = rem & (Psz - 1);
  int bh  = rem >> 11;              // P = 2048 = 2^11
  cache[((size_t)bh * Tt + p) * DHd + d] = (__bf16)past[idx];
}

// ---------------- GEMM: out = A(bf16 [M,K]) * W(bf16 [N,K])^T ---------
// mode 0: write f32 [M,N] to outF (final projection -> d_out)
// mode 1: write bf16 head-split [B,H,dstSeq,DH] at seq offset seqOff
constexpr int MBLK = 128, NBLK = 128, KBLK = 32;
constexpr int GLD  = KBLK + 8;      // padded LDS stride (elements) = 40
constexpr int GTILE = MBLK * GLD;   // elements per LDS tile
// TDM pad: rows of 32 elems = 16 dwords, pad +4 dwords -> interval 3, amount 3

__global__ __launch_bounds__(256)
void gemm_xwt(const __bf16* __restrict__ A, const __bf16* __restrict__ W,
              float* __restrict__ outF, __bf16* __restrict__ outH,
              int mode, int dstSeq, int seqOff) {
  // double-buffered: [A0 | A1 | W0 | W1]  (pointers derived arithmetically --
  // no pointer arrays: addrspacecast static initializers break ld.lld)
  __shared__ __attribute__((aligned(16))) __bf16 smem[4 * GTILE];

  const int t     = threadIdx.x;
  const int wid   = t >> 5;
  const int swid  = __builtin_amdgcn_readfirstlane(wid);  // scalar wave id
  const int lane  = t & 31;
  const int mbase = blockIdx.x * MBLK;
  const int nbase = blockIdx.y * NBLK;
  const int K     = Esz;

  v8f acc[8];
  const v8f zero = {0.f, 0.f, 0.f, 0.f, 0.f, 0.f, 0.f, 0.f};
#pragma unroll
  for (int j = 0; j < 8; ++j) acc[j] = zero;

  // prologue: DMA first K-chunk into buffer 0 (wave0: A, wave1: W)
  if (swid == 0)
    tdm_load_2d(0u, A + (size_t)mbase * K, MBLK, KBLK, K, 3, 3);
  else if (swid == 1)
    tdm_load_2d((uint32_t)(2 * GTILE * sizeof(__bf16)),
                W + (size_t)nbase * K, NBLK, KBLK, K, 3, 3);

  int ib = 0;
  for (int kc = 0; kc < K; kc += KBLK, ib ^= 1) {
    if (swid <= 1) __builtin_amdgcn_s_wait_tensorcnt(0);  // buffer ib ready
    __syncthreads();
    lds_publish(smem);

    // overlap: DMA next chunk into the other buffer while we compute
    if (kc + KBLK < K) {
      const uint32_t nb = (uint32_t)(ib ^ 1) * (uint32_t)(GTILE * sizeof(__bf16));
      if (swid == 0)
        tdm_load_2d(nb, A + (size_t)mbase * K + kc + KBLK, MBLK, KBLK, K, 3, 3);
      else if (swid == 1)
        tdm_load_2d((uint32_t)(2 * GTILE * sizeof(__bf16)) + nb,
                    W + (size_t)nbase * K + kc + KBLK, NBLK, KBLK, K, 3, 3);
    }

    const __bf16* At = smem + ib * GTILE;
    const __bf16* Wt = smem + 2 * GTILE + ib * GTILE;

    v16bf a = load_A_frag(&At[(wid * 16) * GLD], GLD);
#pragma unroll
    for (int j = 0; j < 8; ++j) {
      v16bf b = load_B_frag_nk(&Wt[(j * 16) * GLD], GLD);
      acc[j] = __builtin_amdgcn_wmma_f32_16x16x32_bf16(
          false, a, false, b, (short)0, acc[j], false, false);
    }
    // next iteration's barrier orders these reads before the DMA that
    // overwrites this buffer (two iterations from now).
  }

  // store (C layout: row = r + 8*half, col = lane&15)
  const int n_in = lane & 15;
  const int half = lane >> 4;
#pragma unroll
  for (int j = 0; j < 8; ++j) {
#pragma unroll
    for (int r = 0; r < 8; ++r) {
      const int m = mbase + wid * 16 + r + half * 8;
      const int n = nbase + j * 16 + n_in;
      const float v = acc[j][r];
      if (mode == 0) {
        outF[(size_t)m * Esz + n] = v;
      } else {
        const int b  = m / Ssz, s = m % Ssz;
        const int h  = n >> 6,  d = n & 63;
        outH[(((size_t)(b * Hh + h) * dstSeq) + seqOff + s) * DHd + d] = (__bf16)v;
      }
    }
  }
}

// ---------------- flash attention (bf16 WMMA, f32 softmax) ------------
constexpr int KLD = DHd + 8;    // 72, padded stride for K/V tiles
constexpr int PLD = 32 + 8;     // 40, padded stride for P tiles
constexpr int KVT = 32 * KLD;   // elements per K or V tile
// TDM pad: rows of 64 elems = 32 dwords, pad +4 dwords -> interval 4, amount 3

__global__ __launch_bounds__(128)
void flash_attn(const __bf16* __restrict__ Qh, const __bf16* __restrict__ Kc,
                const __bf16* __restrict__ Vc, __bf16* __restrict__ Oh) {
  // [K0 | K1 | V0 | V1 | P(4 waves)] -- pointers derived arithmetically
  __shared__ __attribute__((aligned(16))) __bf16 smem[4 * KVT + 4 * 16 * PLD];

  const int t     = threadIdx.x;
  const int wid   = t >> 5;
  const int swid  = __builtin_amdgcn_readfirstlane(wid);
  const int lane  = t & 31;
  const int n_in  = lane & 15;
  const int half  = lane >> 4;
  const int qblk  = blockIdx.x;          // 0..S/64-1
  const int bh    = blockIdx.y;          // 0..B*H-1
  const int q0    = qblk * 64;
  const int qbase = q0 + wid * 16;       // this wave's first query row

  const __bf16* qptr = Qh + ((size_t)bh * Ssz + qbase) * DHd;
  const v16bf qa0 = load_A_frag(qptr, DHd);        // dh 0..31
  const v16bf qa1 = load_A_frag(qptr + 32, DHd);   // dh 32..63

  float rmax[8], rsum[8];
  v8f o[4];
  const v8f zero = {0.f, 0.f, 0.f, 0.f, 0.f, 0.f, 0.f, 0.f};
#pragma unroll
  for (int r = 0; r < 8; ++r) { rmax[r] = -3.0e38f; rsum[r] = 0.f; }
#pragma unroll
  for (int nc = 0; nc < 4; ++nc) o[nc] = zero;

  const int ntiles = (Psz + q0 + 64) / 32;          // causal extent of this block
  const float scl = 0.125f;                          // 1/sqrt(DH)
  const __bf16* kbase_p = Kc + (size_t)bh * Tt * DHd;
  const __bf16* vbase_p = Vc + (size_t)bh * Tt * DHd;

  // prologue: DMA tile 0 into buffer 0 (wave0: K, wave1: V)
  if (swid == 0)
    tdm_load_2d(0u, kbase_p, 32, DHd, DHd, 4, 3);
  else if (swid == 1)
    tdm_load_2d((uint32_t)(2 * KVT * sizeof(__bf16)), vbase_p, 32, DHd, DHd, 4, 3);

  int ib = 0;
  for (int kt = 0; kt < ntiles; ++kt, ib ^= 1) {
    const int jb = kt * 32;

    if (swid <= 1) __builtin_amdgcn_s_wait_tensorcnt(0);  // buffer ib ready
    __syncthreads();
    lds_publish(smem);

    // overlap: DMA tile kt+1 into the other buffer while we compute
    if (kt + 1 < ntiles) {
      const size_t nxt = (size_t)(jb + 32) * DHd;
      const uint32_t nb = (uint32_t)(ib ^ 1) * (uint32_t)(KVT * sizeof(__bf16));
      if (swid == 0)
        tdm_load_2d(nb, kbase_p + nxt, 32, DHd, DHd, 4, 3);
      else if (swid == 1)
        tdm_load_2d((uint32_t)(2 * KVT * sizeof(__bf16)) + nb,
                    vbase_p + nxt, 32, DHd, DHd, 4, 3);
    }

    const __bf16* Ktile = smem + ib * KVT;
    const __bf16* Vtile = smem + 2 * KVT + ib * KVT;
    __bf16* Pt          = smem + 4 * KVT;

    const bool active = (jb <= Psz + qbase + 15);   // wave-uniform
    if (active) {
      // ---- scores: two 16x16 tiles (keys jb..+15, jb+16..+31) ----
      v8f c0 = zero, c1 = zero;
      {
        v16bf b00 = load_B_frag_nk(&Ktile[0 * KLD + 0],  KLD);
        v16bf b01 = load_B_frag_nk(&Ktile[0 * KLD + 32], KLD);
        c0 = __builtin_amdgcn_wmma_f32_16x16x32_bf16(false, qa0, false, b00, (short)0, c0, false, false);
        c0 = __builtin_amdgcn_wmma_f32_16x16x32_bf16(false, qa1, false, b01, (short)0, c0, false, false);
        v16bf b10 = load_B_frag_nk(&Ktile[16 * KLD + 0],  KLD);
        v16bf b11 = load_B_frag_nk(&Ktile[16 * KLD + 32], KLD);
        c1 = __builtin_amdgcn_wmma_f32_16x16x32_bf16(false, qa0, false, b10, (short)0, c1, false, false);
        c1 = __builtin_amdgcn_wmma_f32_16x16x32_bf16(false, qa1, false, b11, (short)0, c1, false, false);
      }

      // ---- scale + causal mask + online softmax ----
      const int j0 = jb + n_in;
      const int j1 = jb + 16 + n_in;
#pragma unroll
      for (int r = 0; r < 8; ++r) {
        const int i_abs = qbase + r + half * 8;
        const int limit = i_abs + Psz;
        float s0 = c0[r] * scl; if (j0 > limit) s0 = -1.0e30f;
        float s1 = c1[r] * scl; if (j1 > limit) s1 = -1.0e30f;

        float tm = fmaxf(s0, s1);
#pragma unroll
        for (int m = 1; m <= 8; m <<= 1) tm = fmaxf(tm, __shfl_xor(tm, m, 32));

        const float nm = fmaxf(rmax[r], tm);
        const float sc = __expf(rmax[r] - nm);
        rmax[r] = nm;

        const float p0 = __expf(s0 - nm);
        const float p1 = __expf(s1 - nm);
        float ps = p0 + p1;
#pragma unroll
        for (int m = 1; m <= 8; m <<= 1) ps += __shfl_xor(ps, m, 32);
        rsum[r] = rsum[r] * sc + ps;
#pragma unroll
        for (int nc = 0; nc < 4; ++nc) o[nc][r] *= sc;

        // re-stripe P (C layout -> [16 rows][32 keys]); wave-private region,
        // same-wave LDS ops are in-order: no barrier needed before reload.
        const int row = r + half * 8;
        Pt[wid * 16 * PLD + row * PLD + n_in]      = (__bf16)p0;
        Pt[wid * 16 * PLD + row * PLD + 16 + n_in] = (__bf16)p1;
      }

      // ---- P x V ----
      v16bf pa = load_A_frag(&Pt[wid * 16 * PLD], PLD);
#pragma unroll
      for (int nc = 0; nc < 4; ++nc) {
        v16bf vb = load_B_frag_kn(&Vtile[nc * 16], KLD);
        o[nc] = __builtin_amdgcn_wmma_f32_16x16x32_bf16(
            false, pa, false, vb, (short)0, o[nc], false, false);
      }
    }
    // next iteration's barrier orders these reads before the DMA that
    // overwrites this buffer (two iterations from now).
  }

  // ---- normalize + store bf16 [B*S, E] (col = h*64 + d) ----
  const int b = bh / Hh;
  const int h = bh % Hh;
#pragma unroll
  for (int r = 0; r < 8; ++r) {
    const float inv = 1.0f / rsum[r];
    const int s = qbase + r + half * 8;
#pragma unroll
    for (int nc = 0; nc < 4; ++nc) {
      Oh[(size_t)(b * Ssz + s) * Esz + h * 64 + nc * 16 + n_in] =
          (__bf16)(o[nc][r] * inv);
    }
  }
}

// ---------------- host-side launcher ----------------------------------
extern "C" void kernel_launch(void* const* d_in, const int* in_sizes, int n_in,
                              void* d_out, int out_size, void* d_ws, size_t ws_size,
                              hipStream_t stream) {
  const float* query  = (const float*)d_in[0];
  const float* key_in = (const float*)d_in[1];
  const float* value  = (const float*)d_in[2];
  const float* past_k = (const float*)d_in[3];
  const float* past_v = (const float*)d_in[4];
  // d_in[5] = attn_mask (causal; recomputed on device)
  const float* Wq = (const float*)d_in[6];
  const float* Wk = (const float*)d_in[7];
  const float* Wv = (const float*)d_in[8];
  const float* Wo = (const float*)d_in[9];

  // workspace layout (bf16 buffers), 256B aligned
  char* ws = (char*)d_ws;
  size_t off = 0;
  auto alloc = [&](size_t bytes) {
    size_t o = off;
    off = (off + bytes + 255) & ~(size_t)255;
    return o;
  };
  const size_t actB = (size_t)Mrows * Esz * sizeof(__bf16);          // 8 MB
  const size_t wB   = (size_t)Esz * Esz * sizeof(__bf16);            // 2 MB
  const size_t qhB  = (size_t)Bsz * Hh * Ssz * DHd * sizeof(__bf16); // 8 MB
  const size_t kvB  = (size_t)Bsz * Hh * Tt * DHd * sizeof(__bf16);  // 16 MB

  __bf16* bfQ  = (__bf16*)(ws + alloc(actB));
  __bf16* bfK  = (__bf16*)(ws + alloc(actB));
  __bf16* bfV  = (__bf16*)(ws + alloc(actB));
  __bf16* bfWq = (__bf16*)(ws + alloc(wB));
  __bf16* bfWk = (__bf16*)(ws + alloc(wB));
  __bf16* bfWv = (__bf16*)(ws + alloc(wB));
  __bf16* bfWo = (__bf16*)(ws + alloc(wB));
  __bf16* Qh   = (__bf16*)(ws + alloc(qhB));
  __bf16* Kc   = (__bf16*)(ws + alloc(kvB));
  __bf16* Vc   = (__bf16*)(ws + alloc(kvB));
  __bf16* Ao   = (__bf16*)(ws + alloc(actB));
  (void)ws_size; (void)n_in; (void)in_sizes; (void)out_size;

  const int nAct = Mrows * Esz;          // 4,194,304
  const int nW   = Esz * Esz;            // 1,048,576
  const int nPKV = Bsz * Hh * Psz * DHd; // 4,194,304

  dim3 blk(256);
  cast_f32_to_bf16<<<(nAct + 255) / 256, blk, 0, stream>>>(query,  bfQ, nAct);
  cast_f32_to_bf16<<<(nAct + 255) / 256, blk, 0, stream>>>(key_in, bfK, nAct);
  cast_f32_to_bf16<<<(nAct + 255) / 256, blk, 0, stream>>>(value,  bfV, nAct);
  cast_f32_to_bf16<<<(nW + 255) / 256,   blk, 0, stream>>>(Wq, bfWq, nW);
  cast_f32_to_bf16<<<(nW + 255) / 256,   blk, 0, stream>>>(Wk, bfWk, nW);
  cast_f32_to_bf16<<<(nW + 255) / 256,   blk, 0, stream>>>(Wv, bfWv, nW);
  cast_f32_to_bf16<<<(nW + 255) / 256,   blk, 0, stream>>>(Wo, bfWo, nW);
  scatter_past<<<(nPKV + 255) / 256, blk, 0, stream>>>(past_k, Kc, nPKV);
  scatter_past<<<(nPKV + 255) / 256, blk, 0, stream>>>(past_v, Vc, nPKV);

  dim3 ggrid(Mrows / MBLK, Esz / NBLK);  // (32, 8)
  gemm_xwt<<<ggrid, 256, 0, stream>>>(bfQ, bfWq, nullptr, Qh, 1, Ssz, 0);
  gemm_xwt<<<ggrid, 256, 0, stream>>>(bfK, bfWk, nullptr, Kc, 1, Tt, Psz);
  gemm_xwt<<<ggrid, 256, 0, stream>>>(bfV, bfWv, nullptr, Vc, 1, Tt, Psz);

  dim3 agrid(Ssz / 64, Bsz * Hh);        // (32, 32)
  flash_attn<<<agrid, 128, 0, stream>>>(Qh, Kc, Vc, Ao);

  gemm_xwt<<<ggrid, 256, 0, stream>>>(Ao, bfWo, (float*)d_out, nullptr, 0, 0, 0);
}
